// MLDecoder_17076789969007
// MI455X (gfx1250) — matrix operand order repaired and checked by reading, not compiled, not run on hardware
//
#include <hip/hip_runtime.h>

typedef _Float16 f16;
typedef __attribute__((ext_vector_type(16))) _Float16 v16h;
typedef __attribute__((ext_vector_type(8)))  float    v8f;
typedef __attribute__((ext_vector_type(4)))  unsigned int u32x4;

union Frag { v16h h; u32x4 u[2]; };

#define BM 128
#define BN 128
#define BKK 32
#define SAK 40   // padded LDS row stride in halves (80B rows -> 16B aligned frag loads)

// ---------------------------------------------------------------------------
// Generic f16 WMMA GEMM:  C[M,N] = act( A[M,K](f16,lda) * Bt[N,K]^T + bias )
// Bt is the transposed (row-major [N][K]) f16 weight copy. Double-buffered LDS,
// one barrier per K-step; OOB rows/cols handled by index clamping (their C
// elements are simply never stored), so the hot loop has no exec-mask churn.
// ---------------------------------------------------------------------------
__global__ __launch_bounds__(256) void gemm_f16_kernel(
    const f16* __restrict__ A, long lda, long bsA,
    const f16* __restrict__ Bt, long ldb, long bsB,
    const float* __restrict__ bias, long bsBias,
    float* __restrict__ o32, long ldo32, long bsO32,
    f16* __restrict__ o16, long ldo16, long bsO16,
    int M, int N, int K, int relu)
{
    __shared__ __align__(16) f16 sA[2][BM * SAK];
    __shared__ __align__(16) f16 sB[2][BN * SAK];

    const long bz = blockIdx.z;
    A  += bz * bsA;
    Bt += bz * bsB;
    const float* biasP = bias ? bias + bz * bsBias : nullptr;
    float* o32P = o32 ? o32 + bz * bsO32 : nullptr;
    f16*   o16P = o16 ? o16 + bz * bsO16 : nullptr;

    const int tid  = threadIdx.x;
    const int wave = tid >> 5;
    const int lane = tid & 31;
    const int wm = wave >> 1;      // 0..3  -> 32-row slab
    const int wn = wave & 1;       // 0..1  -> 64-col slab
    const int row0 = blockIdx.y * BM;
    const int col0 = blockIdx.x * BN;

    v8f c[2][4];
    #pragma unroll
    for (int fm = 0; fm < 2; ++fm)
        #pragma unroll
        for (int fn = 0; fn < 4; ++fn)
            c[fm][fn] = (v8f){0.f,0.f,0.f,0.f,0.f,0.f,0.f,0.f};

    const int lm = tid >> 1;            // 0..127: tile row/col loaded by this thread
    const int lk = (tid & 1) * 16;      // 0 or 16: K-chunk within BK=32

    // clamped source row/col: always in-bounds; surplus rows/cols only feed
    // C elements that the epilogue masks off.
    int gr = row0 + lm; if (gr > M - 1) gr = M - 1;
    int gc = col0 + lm; if (gc > N - 1) gc = N - 1;
    const f16* pa = A  + (long)gr * lda + lk;
    const f16* pb = Bt + (long)gc * ldb + lk;

    // fragment LDS addresses (within a buffer)
    const int l15 = lane & 15;
    const int kg  = lane >> 4;
    const int aOff = (wm * 32 + l15) * SAK + kg * 8;     // + fm*16*SAK
    const int bOff = (wn * 64 + l15) * SAK + kg * 8;     // + fn*16*SAK
    const int stOffA = lm * SAK + lk;

    // ---- preload tile 0 ----
    u32x4 ra0 = *(const u32x4*)pa, ra1 = *(const u32x4*)(pa + 8);
    u32x4 rb0 = *(const u32x4*)pb, rb1 = *(const u32x4*)(pb + 8);
    *(u32x4*)&sA[0][stOffA]     = ra0;
    *(u32x4*)&sA[0][stOffA + 8] = ra1;
    *(u32x4*)&sB[0][stOffA]     = rb0;
    *(u32x4*)&sB[0][stOffA + 8] = rb1;
    __syncthreads();

    int cur = 0;
    for (int kk = 0; kk < K; kk += BKK) {
        const bool more = (kk + BKK) < K;
        if (more) {   // issue next tile's global loads before this tile's WMMAs
            const f16* qa = pa + kk + BKK;
            const f16* qb = pb + kk + BKK;
            ra0 = *(const u32x4*)qa; ra1 = *(const u32x4*)(qa + 8);
            rb0 = *(const u32x4*)qb; rb1 = *(const u32x4*)(qb + 8);
        }

        // ---- fragments per ISA 7.12.2 + 8 WMMAs ----
        Frag a[2], b[4];
        #pragma unroll
        for (int fm = 0; fm < 2; ++fm) {
            const f16* p = &sA[cur][aOff + fm * 16 * SAK];
            a[fm].u[0] = *(const u32x4*)p;          // K = kg*8 .. +7
            a[fm].u[1] = *(const u32x4*)(p + 16);   // K = 16+kg*8 .. +7
        }
        #pragma unroll
        for (int fn = 0; fn < 4; ++fn) {
            const f16* p = &sB[cur][bOff + fn * 16 * SAK];
            b[fn].u[0] = *(const u32x4*)p;          // K = kg*8 .. +7
            b[fn].u[1] = *(const u32x4*)(p + 16);   // K = 16+kg*8 .. +7
        }
        #pragma unroll
        for (int fm = 0; fm < 2; ++fm)
            #pragma unroll
            for (int fn = 0; fn < 4; ++fn)
                c[fm][fn] = __builtin_amdgcn_wmma_f32_16x16x32_f16(
                    false, a[fm].h, false, b[fn].h, (short)0, c[fm][fn], false, false);

        if (more) {
            const int nxt = cur ^ 1;
            *(u32x4*)&sA[nxt][stOffA]     = ra0;
            *(u32x4*)&sA[nxt][stOffA + 8] = ra1;
            *(u32x4*)&sB[nxt][stOffA]     = rb0;
            *(u32x4*)&sB[nxt][stOffA + 8] = rb1;
            __syncthreads();                         // one barrier per K-step
            cur = nxt;
        }
    }

    // ---- epilogue: bias + relu + store (C layout: lane0-15 N=lane M=r, lane16-31 M=8+r) ----
    const int hiM = (lane >> 4) * 8;
    #pragma unroll
    for (int fm = 0; fm < 2; ++fm)
        #pragma unroll
        for (int fn = 0; fn < 4; ++fn) {
            int col = col0 + wn * 64 + fn * 16 + l15;
            if (col >= N) continue;
            float bv = biasP ? biasP[col] : 0.f;
            #pragma unroll
            for (int r = 0; r < 8; ++r) {
                int row = row0 + wm * 32 + fm * 16 + hiM + r;
                if (row >= M) continue;
                float v = c[fm][fn][r] + bv;
                if (relu) v = fmaxf(v, 0.f);
                if (o32P) o32P[(long)row * ldo32 + col] = v;
                if (o16P) o16P[(long)row * ldo16 + col] = (f16)v;
            }
        }
}

// ---------------------------------------------------------------------------
// Transpose + f32->f16 convert: out[c][r] = in[r][colStart+c]; out stride = R
// ---------------------------------------------------------------------------
__global__ void transpose_f16_kernel(const float* __restrict__ in, long inBS,
                                     f16* __restrict__ out, long outBS,
                                     int R, int Ctot, int colStart, int Ccnt)
{
    __shared__ float tile[32][33];
    const long bz = blockIdx.z;
    in  += bz * inBS;
    out += bz * outBS;
    int c0 = blockIdx.x * 32, r0 = blockIdx.y * 32;
    int tx = threadIdx.x, ty = threadIdx.y;    // (32, 8)
    #pragma unroll
    for (int i = 0; i < 32; i += 8) {
        int r = r0 + ty + i, cc = c0 + tx;
        if (r < R && cc < Ccnt) tile[ty + i][tx] = in[(long)r * Ctot + colStart + cc];
    }
    __syncthreads();
    #pragma unroll
    for (int i = 0; i < 32; i += 8) {
        int cc = c0 + ty + i, r = r0 + tx;
        if (cc < Ccnt && r < R) out[(long)cc * R + r] = (f16)tile[tx][ty + i];
    }
}

__global__ void convert_f16_kernel(const float* __restrict__ in, f16* __restrict__ out, long n)
{
    long i = (long)blockIdx.x * blockDim.x + threadIdx.x;
    long stride = (long)gridDim.x * blockDim.x;
    for (; i < n; i += stride) out[i] = (f16)in[i];
}

// ---------------------------------------------------------------------------
// Stage A (batch-independent): tgt = LN1(2*query_embed); q = (tgt@Wq + bq)/sqrt(96)
// One block per query group g.
// ---------------------------------------------------------------------------
__global__ __launch_bounds__(256) void stageA_kernel(
    const float* __restrict__ qe, const float* __restrict__ ln1g, const float* __restrict__ ln1b,
    const float* __restrict__ w_qkv, const float* __restrict__ b_qkv,
    float* __restrict__ tgt_out, f16* __restrict__ q_out)
{
    __shared__ float t[768];
    __shared__ float rbuf[256];
    const int g = blockIdx.x, tid = threadIdx.x;

    float v[3]; float s = 0.f;
    #pragma unroll
    for (int p = 0; p < 3; ++p) { int d = tid + p * 256; v[p] = 2.f * qe[g * 768 + d]; s += v[p]; }
    rbuf[tid] = s; __syncthreads();
    for (int k = 128; k > 0; k >>= 1) { if (tid < k) rbuf[tid] += rbuf[tid + k]; __syncthreads(); }
    float mean = rbuf[0] * (1.f / 768.f);
    __syncthreads();
    float sv = 0.f;
    #pragma unroll
    for (int p = 0; p < 3; ++p) { float xv = v[p] - mean; sv += xv * xv; }
    rbuf[tid] = sv; __syncthreads();
    for (int k = 128; k > 0; k >>= 1) { if (tid < k) rbuf[tid] += rbuf[tid + k]; __syncthreads(); }
    float inv = rsqrtf(rbuf[0] * (1.f / 768.f) + 1e-5f);
    __syncthreads();
    #pragma unroll
    for (int p = 0; p < 3; ++p) {
        int d = tid + p * 256;
        float y = (v[p] - mean) * inv * ln1g[d] + ln1b[d];
        t[d] = y;
        tgt_out[g * 768 + d] = y;
    }
    __syncthreads();
    const float scale = 0.1020620726159658f;   // 1/sqrt(96), folded into q
    for (int p = 0; p < 3; ++p) {
        int j = tid + p * 256;
        float acc = b_qkv[j];
        for (int d = 0; d < 768; ++d) acc += t[d] * w_qkv[(long)d * 2304 + j];
        q_out[g * 768 + j] = (f16)(acc * scale);
    }
}

// ---------------------------------------------------------------------------
// LayerNorm over D=768 with fused residual: y = LN(in[row] + res[row % resMod])
// One block per row; optional f32 and f16 outputs.
// ---------------------------------------------------------------------------
__global__ __launch_bounds__(256) void ln_kernel(
    const float* __restrict__ in, const float* __restrict__ res, int resMod,
    const float* __restrict__ gam, const float* __restrict__ bet,
    float* __restrict__ out32, f16* __restrict__ out16)
{
    __shared__ float rbuf[256];
    const long row = blockIdx.x;
    const int tid = threadIdx.x;
    const float* ip = in + row * 768;
    const float* rp = res + (long)(row % (long)resMod) * 768;

    float v[3]; float s = 0.f;
    #pragma unroll
    for (int p = 0; p < 3; ++p) { int d = tid + p * 256; v[p] = ip[d] + rp[d]; s += v[p]; }
    rbuf[tid] = s; __syncthreads();
    for (int k = 128; k > 0; k >>= 1) { if (tid < k) rbuf[tid] += rbuf[tid + k]; __syncthreads(); }
    float mean = rbuf[0] * (1.f / 768.f);
    __syncthreads();
    float sv = 0.f;
    #pragma unroll
    for (int p = 0; p < 3; ++p) { float xv = v[p] - mean; sv += xv * xv; }
    rbuf[tid] = sv; __syncthreads();
    for (int k = 128; k > 0; k >>= 1) { if (tid < k) rbuf[tid] += rbuf[tid + k]; __syncthreads(); }
    float inv = rsqrtf(rbuf[0] * (1.f / 768.f) + 1e-5f);
    #pragma unroll
    for (int p = 0; p < 3; ++p) {
        int d = tid + p * 256;
        float y = (v[p] - mean) * inv * gam[d] + bet[d];
        if (out32) out32[row * 768 + d] = y;
        if (out16) out16[row * 768 + d] = (f16)y;
    }
}

// ---------------------------------------------------------------------------
// Cross-attention: per (b,h): scores = q(100x96) k^T(96x49), softmax, o = p v.
// q already scaled by 1/sqrt(96). One block per (b,h), 128 threads.
// ---------------------------------------------------------------------------
__global__ __launch_bounds__(128) void attention_kernel(
    const f16* __restrict__ q, const f16* __restrict__ k,
    const f16* __restrict__ v, f16* __restrict__ o)
{
    __shared__ f16 sq[100][96];
    __shared__ f16 sk[49][96];
    __shared__ f16 sv[49][96];
    __shared__ float ssc[128][50];

    const int bh = blockIdx.x;
    const int b = bh >> 3, h = bh & 7;
    const int tid = threadIdx.x;

    for (int i = tid; i < 100 * 96; i += 128) {
        int g = i / 96, d = i % 96;
        sq[g][d] = q[(long)g * 768 + h * 96 + d];
    }
    for (int i = tid; i < 49 * 96; i += 128) {
        int s = i / 96, d = i % 96;
        long gi = ((long)(b * 49 + s)) * 768 + h * 96 + d;
        sk[s][d] = k[gi];
        sv[s][d] = v[gi];
    }
    __syncthreads();

    if (tid < 100) {
        const int g = tid;
        float mx = -1e30f;
        for (int s = 0; s < 49; ++s) {
            float acc = 0.f;
            #pragma unroll 8
            for (int d = 0; d < 96; ++d) acc += (float)sq[g][d] * (float)sk[s][d];
            ssc[tid][s] = acc;
            mx = fmaxf(mx, acc);
        }
        float denom = 0.f;
        for (int s = 0; s < 49; ++s) { float e = __expf(ssc[tid][s] - mx); ssc[tid][s] = e; denom += e; }
        float rd = 1.f / denom;
        f16* op = o + ((long)(b * 100 + g)) * 768 + h * 96;
        for (int d = 0; d < 96; ++d) {
            float acc = 0.f;
            for (int s = 0; s < 49; ++s) acc += ssc[tid][s] * (float)sv[s][d];
            op[d] = (f16)(acc * rd);
        }
    }
}

// ---------------------------------------------------------------------------
extern "C" void kernel_launch(void* const* d_in, const int* in_sizes, int n_in,
                              void* d_out, int out_size, void* d_ws, size_t ws_size,
                              hipStream_t stream)
{
    (void)in_sizes; (void)n_in; (void)out_size;
    const float* x          = (const float*)d_in[0];
    const float* w_embed    = (const float*)d_in[1];
    const float* b_embed    = (const float*)d_in[2];
    const float* query_emb  = (const float*)d_in[3];
    const float* w_qkv      = (const float*)d_in[4];
    const float* b_qkv      = (const float*)d_in[5];
    const float* w_attn_out = (const float*)d_in[6];
    const float* b_attn_out = (const float*)d_in[7];
    const float* ln1_g = (const float*)d_in[8],  *ln1_b = (const float*)d_in[9];
    const float* ln2_g = (const float*)d_in[10], *ln2_b = (const float*)d_in[11];
    const float* ln3_g = (const float*)d_in[12], *ln3_b = (const float*)d_in[13];
    const float* w1 = (const float*)d_in[14], *b1 = (const float*)d_in[15];
    const float* w2 = (const float*)d_in[16], *b2 = (const float*)d_in[17];
    const float* dup_pool = (const float*)d_in[18];
    const float* dup_bias = (const float*)d_in[19];
    float* logits = (float*)d_out;

    const int Bn = 256, S = 49, C0 = 2048, D = 768, Ff = 2048, G = 100, DF = 96, NC = 9600;
    const int M1 = Bn * S;     // 12544
    const int M2 = Bn * G;     // 25600

    char* ws = (char*)d_ws;
    size_t off = 0;
    auto alloc = [&](size_t bytes) { size_t o = off; off += (bytes + 255) & ~(size_t)255; return o; };

    f16* xf16    = (f16*)(ws + alloc((size_t)M1 * C0 * 2));   // also reused as o_f16 later
    f16* wembT   = (f16*)(ws + alloc((size_t)D * C0 * 2));    // [D][C0]
    f16* wkT     = (f16*)(ws + alloc((size_t)D * D * 2));
    f16* wvT     = (f16*)(ws + alloc((size_t)D * D * 2));
    f16* wattnT  = (f16*)(ws + alloc((size_t)D * D * 2));
    f16* w1T     = (f16*)(ws + alloc((size_t)Ff * D * 2));    // [F][D]
    f16* w2T     = (f16*)(ws + alloc((size_t)D * Ff * 2));    // [D][F]
    f16* dupT    = (f16*)(ws + alloc((size_t)G * DF * D * 2));// per g: [DF][D]
    f16* memf16  = (f16*)(ws + alloc((size_t)M1 * D * 2));    // start of mem/k/v region
    f16* kf16    = (f16*)(ws + alloc((size_t)M1 * D * 2));
    f16* vf16    = (f16*)(ws + alloc((size_t)M1 * D * 2));
    float* tgtF  = (float*)(ws + alloc((size_t)G * D * 4));
    f16* qf16    = (f16*)(ws + alloc((size_t)G * D * 2));
    float* s1    = (float*)(ws + alloc((size_t)M2 * D * 4));  // attn-proj out, later ff2 out
    float* tgt2F = (float*)(ws + alloc((size_t)M2 * D * 4));
    f16* tgt2h   = (f16*)(ws + alloc((size_t)M2 * D * 2));
    f16* ff1h    = (f16*)(ws + alloc((size_t)M2 * Ff * 2));
    f16* of16 = xf16;      // x dead after embed GEMM  (needs 39.3MB <= 51.4MB)
    f16* hf16 = memf16;    // mem/k/v dead after attn  (needs 39.3MB <= 57.8MB)
    if (off > ws_size) return;

    dim3 tb(32, 8);
    auto tgrid = [](int Ccnt, int R, int z) { return dim3((Ccnt + 31) / 32, (R + 31) / 32, z); };

    // 1) weight transposes -> f16 [N][K]
    transpose_f16_kernel<<<tgrid(D, C0, 1), tb, 0, stream>>>(w_embed, 0, wembT, 0, C0, D, 0, D);
    transpose_f16_kernel<<<tgrid(D, D, 1), tb, 0, stream>>>(w_qkv, 0, wkT, 0, D, 3 * D, D, D);
    transpose_f16_kernel<<<tgrid(D, D, 1), tb, 0, stream>>>(w_qkv, 0, wvT, 0, D, 3 * D, 2 * D, D);
    transpose_f16_kernel<<<tgrid(D, D, 1), tb, 0, stream>>>(w_attn_out, 0, wattnT, 0, D, D, 0, D);
    transpose_f16_kernel<<<tgrid(Ff, D, 1), tb, 0, stream>>>(w1, 0, w1T, 0, D, Ff, 0, Ff);
    transpose_f16_kernel<<<tgrid(D, Ff, 1), tb, 0, stream>>>(w2, 0, w2T, 0, Ff, D, 0, D);
    transpose_f16_kernel<<<tgrid(DF, D, G), tb, 0, stream>>>(dup_pool, (long)D * DF, dupT, (long)DF * D, D, DF, 0, DF);

    // 2) x -> f16
    convert_f16_kernel<<<4096, 256, 0, stream>>>(x, xf16, (long)M1 * C0);

    // 3) tgt = LN1(2*query_embed); q = (tgt@Wq+bq)/sqrt(HD)
    stageA_kernel<<<G, 256, 0, stream>>>(query_emb, ln1_g, ln1_b, w_qkv, b_qkv, tgtF, qf16);

    auto gemm = [&](const f16* A, long lda, long bsA, const f16* Bt, long ldb, long bsB,
                    const float* bias, long bsBias,
                    float* o32, long ldo32, long bsO32,
                    f16* o16, long ldo16, long bsO16,
                    int M, int N, int K, int relu, int batch) {
        dim3 grid((N + BN - 1) / BN, (M + BM - 1) / BM, batch);
        gemm_f16_kernel<<<grid, 256, 0, stream>>>(A, lda, bsA, Bt, ldb, bsB, bias, bsBias,
                                                  o32, ldo32, bsO32, o16, ldo16, bsO16, M, N, K, relu);
    };

    // 4) mem = relu(x @ w_embed + b_embed)  -> f16 [M1,D]
    gemm(xf16, C0, 0, wembT, C0, 0, b_embed, 0, nullptr, 0, 0, memf16, D, 0, M1, D, C0, 1, 1);
    // 5) k = mem @ Wk + bk ; v = mem @ Wv + bv
    gemm(memf16, D, 0, wkT, D, 0, b_qkv + D, 0, nullptr, 0, 0, kf16, D, 0, M1, D, D, 0, 1);
    gemm(memf16, D, 0, wvT, D, 0, b_qkv + 2 * D, 0, nullptr, 0, 0, vf16, D, 0, M1, D, D, 0, 1);
    // 6) attention -> o_f16 [B,G,D]
    attention_kernel<<<Bn * 8, 128, 0, stream>>>(qf16, kf16, vf16, of16);
    // 7) s1 = o @ w_attn_out + b_attn_out  (f32)
    gemm(of16, D, 0, wattnT, D, 0, b_attn_out, 0, s1, D, 0, nullptr, 0, 0, M2, D, D, 0, 1);
    // 8) tgt2 = LN2(tgt + s1)  (residual broadcast over batch via row % G)
    ln_kernel<<<M2, 256, 0, stream>>>(s1, tgtF, G, ln2_g, ln2_b, tgt2F, tgt2h);
    // 9) ff1 = relu(tgt2 @ w1 + b1) -> f16 [M2,F]
    gemm(tgt2h, D, 0, w1T, D, 0, b1, 0, nullptr, 0, 0, ff1h, Ff, 0, M2, Ff, D, 1, 1);
    // 10) ff2 = ff1 @ w2 + b2 -> f32 (reuse s1)
    gemm(ff1h, Ff, 0, w2T, Ff, 0, b2, 0, s1, D, 0, nullptr, 0, 0, M2, D, Ff, 0, 1);
    // 11) h = LN3(tgt2 + ff2) -> f16
    ln_kernel<<<M2, 256, 0, stream>>>(s1, tgt2F, M2, ln3_g, ln3_b, nullptr, hf16);
    // 12) GroupFC: per g: logits[:, g*96:(g+1)*96] = h[:,g,:] @ dup_pool[g] + dup_bias[g*96:]
    gemm(hf16, (long)G * D, D, dupT, D, (long)DF * D, dup_bias, DF,
         logits, NC, DF, nullptr, 0, 0, Bn, DF, D, 0, G);
}
